// RandomTreeCell_31258771981062
// MI455X (gfx1250) — matrix operand order repaired and checked by reading, not compiled, not run on hardware
//
#include <hip/hip_runtime.h>
#include <hip/hip_bf16.h>
#include <math.h>

// ---------------------------------------------------------------------------
// Types for CDNA5 WMMA (wave32): v_wmma_f32_16x16x32_bf16
// ---------------------------------------------------------------------------
typedef __attribute__((ext_vector_type(16))) __bf16 v16bf;
typedef __attribute__((ext_vector_type(8)))  float  v8f;

#define BM 64
#define BN 128
#define BK 64
#define LDK 72   // BK + 8 bf16 pad -> 144B row stride (16B aligned, bank-spread)

union FragAB { uint4 u[2]; v16bf v; };

__device__ __forceinline__ float sigmoidf_(float x) { return 1.0f / (1.0f + expf(-x)); }
__device__ __forceinline__ float geluf_(float x)    { return 0.5f * x * (1.0f + erff(x * 0.70710678118654752f)); }

// ---------------------------------------------------------------------------
// CDNA5 async global->LDS copy (ASYNCcnt-tracked, no VGPR staging).
// Builtin signature (from hipcc diagnostic): param0 = v4i __device__* (AS1),
// param1 = v4i LDS* (AS3), then imm offset, imm cpol.
// Falls back to load+ds_store if the builtin isn't available.
// ---------------------------------------------------------------------------
#if defined(__HIP_DEVICE_COMPILE__) && __has_builtin(__builtin_amdgcn_global_load_async_to_lds_b128) && __has_builtin(__builtin_amdgcn_s_wait_asynccnt)
#define USE_ASYNC_LDS 1
#else
#define USE_ASYNC_LDS 0
#endif

#if USE_ASYNC_LDS
typedef __attribute__((vector_size(16))) int v4i_vs;
typedef __attribute__((address_space(1))) v4i_vs* g_v4i_p;   // global (64-bit ptr)
typedef __attribute__((address_space(3))) v4i_vs* l_v4i_p;   // LDS (32-bit ptr)
__device__ __forceinline__ void async_g2l_b128(const void* g, void* l)
{
    // generic->AS casts via integer (low 32 bits of a generic LDS pointer are
    // the LDS byte offset per the CDNA5 aperture mapping); drops const too.
    __builtin_amdgcn_global_load_async_to_lds_b128(
        (g_v4i_p)(unsigned long long)(size_t)g,
        (l_v4i_p)(unsigned)(size_t)l,
        0, 0);
}
#endif

// ---------------------------------------------------------------------------
// bf16 GEMM: C(M,N) = A(M,K) * B^T(N,K)   (B stored N-major, pre-transposed)
// EPI: 1 = bias + exact GELU -> bf16 out ; 2 = bias -> f32 out
// Block 256 thr = 8 wave32; wave grid 2x4; wave tile 32x32 (2x2 of 16x16).
// No register double-buffer: latency hidden by ~10 resident blocks/WGP
// (27KB LDS each) -> no live-across-barrier staging -> no scratch spills.
// ---------------------------------------------------------------------------
template <int EPI>
__global__ __launch_bounds__(256)
void gemm_bf16_wmma(const __bf16* __restrict__ A, int lda,
                    const __bf16* __restrict__ B, int ldb,
                    const float* __restrict__ bias,
                    void* __restrict__ Cout, int ldc,
                    int M, int K)
{
    __shared__ __align__(16) __bf16 As[BM][LDK];
    __shared__ __align__(16) __bf16 Bs[BN][LDK];

    const int tid   = threadIdx.x;
    const int lane  = tid & 31;
    const int wave  = tid >> 5;
    const int waveM = wave & 1;        // 2 waves along M
    const int waveN = wave >> 1;       // 4 waves along N
    const int row16 = lane & 15;
    const int khi   = lane >> 4;       // lane half select

    const int blockM = blockIdx.x * BM;
    const int blockN = blockIdx.y * BN;

    v8f acc[2][2];
#pragma unroll
    for (int i = 0; i < 2; ++i)
#pragma unroll
        for (int j = 0; j < 2; ++j)
            acc[i][j] = (v8f){0.f, 0.f, 0.f, 0.f, 0.f, 0.f, 0.f, 0.f};

    // per-thread 16B copy chunks: A tile 64x64 bf16 = 512 chunks (2/thr),
    // B tile 128x64 bf16 = 1024 chunks (4/thr)
    int aRow[2], aCol[2], bRow[4], bCol[4];
    const __bf16* aPtr[2];
    const __bf16* bPtr[4];
#pragma unroll
    for (int j = 0; j < 2; ++j) {
        int c = tid + 256 * j; aRow[j] = c >> 3; aCol[j] = (c & 7) * 8;
        int gr = blockM + aRow[j]; gr = (gr < M) ? gr : (M - 1);   // clamp: safe read
        aPtr[j] = A + (size_t)gr * lda + aCol[j];
    }
#pragma unroll
    for (int j = 0; j < 4; ++j) {
        int c = tid + 256 * j; bRow[j] = c >> 3; bCol[j] = (c & 7) * 8;
        bPtr[j] = B + (size_t)(blockN + bRow[j]) * ldb + bCol[j];  // N multiple of 128
    }

    const int nK = K / BK;
    for (int k0 = 0; k0 < nK; ++k0) {
        const int kOff = k0 * BK;
        __syncthreads();
#if USE_ASYNC_LDS
#pragma unroll
        for (int j = 0; j < 2; ++j)
            async_g2l_b128(aPtr[j] + kOff, &As[aRow[j]][aCol[j]]);
#pragma unroll
        for (int j = 0; j < 4; ++j)
            async_g2l_b128(bPtr[j] + kOff, &Bs[bRow[j]][bCol[j]]);
        if (k0 + 1 < nK) {
#pragma unroll
            for (int j = 0; j < 2; ++j) __builtin_prefetch(aPtr[j] + kOff + BK, 0, 1);
#pragma unroll
            for (int j = 0; j < 4; ++j) __builtin_prefetch(bPtr[j] + kOff + BK, 0, 1);
        }
        __builtin_amdgcn_s_wait_asynccnt(0);
#else
#pragma unroll
        for (int j = 0; j < 2; ++j)
            *(uint4*)&As[aRow[j]][aCol[j]] = *(const uint4*)(aPtr[j] + kOff);
#pragma unroll
        for (int j = 0; j < 4; ++j)
            *(uint4*)&Bs[bRow[j]][bCol[j]] = *(const uint4*)(bPtr[j] + kOff);
        if (k0 + 1 < nK) {
#pragma unroll
            for (int j = 0; j < 2; ++j) __builtin_prefetch(aPtr[j] + kOff + BK, 0, 1);
#pragma unroll
            for (int j = 0; j < 4; ++j) __builtin_prefetch(bPtr[j] + kOff + BK, 0, 1);
        }
#endif
        __syncthreads();

#pragma unroll
        for (int s = 0; s < 2; ++s) {          // two K=32 WMMA steps per LDS tile
            FragAB a[2], b[2];
#pragma unroll
            for (int mt = 0; mt < 2; ++mt) {
                const int r  = waveM * 32 + mt * 16 + row16;
                const int kb = s * 32 + khi * 8;   // lanes0-15: K 0..7/16..23 ; lanes16-31: K 8..15/24..31
                a[mt].u[0] = *(const uint4*)&As[r][kb];
                a[mt].u[1] = *(const uint4*)&As[r][kb + 16];
            }
#pragma unroll
            for (int nt = 0; nt < 2; ++nt) {
                const int r  = waveN * 32 + nt * 16 + row16;   // N index (B stored N-major)
                const int kb = s * 32 + khi * 8;               // lanes0-15: K 0..7/16..23 ; lanes16-31: K 8..15/24..31
                b[nt].u[0] = *(const uint4*)&Bs[r][kb];
                b[nt].u[1] = *(const uint4*)&Bs[r][kb + 16];
            }
#pragma unroll
            for (int mt = 0; mt < 2; ++mt)
#pragma unroll
                for (int nt = 0; nt < 2; ++nt)
                    acc[mt][nt] = __builtin_amdgcn_wmma_f32_16x16x32_bf16(
                        false, a[mt].v, false, b[nt].v, (short)0, acc[mt][nt], false, false);
        }
    }

    // Epilogue (C layout: lane0-15 N=lane, M=v ; lane16-31 N=lane-16, M=v+8)
#pragma unroll
    for (int mt = 0; mt < 2; ++mt)
#pragma unroll
        for (int nt = 0; nt < 2; ++nt) {
            const int gr0 = blockM + waveM * 32 + mt * 16 + khi * 8;
            const int gc  = blockN + waveN * 32 + nt * 16 + row16;
            const float bv = bias[gc];
#pragma unroll
            for (int v = 0; v < 8; ++v) {
                const int gr = gr0 + v;
                if (gr < M) {
                    float x = acc[mt][nt][v] + bv;
                    if (EPI == 1) {
                        ((__bf16*)Cout)[(size_t)gr * ldc + gc] = (__bf16)geluf_(x);
                    } else {
                        ((float*)Cout)[(size_t)gr * ldc + gc] = x;
                    }
                }
            }
        }
}

// ---------------------------------------------------------------------------
// Block reduction helper (256 threads, 8 wave32): sum + sumsq
// ---------------------------------------------------------------------------
__device__ __forceinline__ void blockReduce2(float& s, float& q, float* shm)
{
#pragma unroll
    for (int o = 16; o; o >>= 1) { s += __shfl_xor(s, o, 32); q += __shfl_xor(q, o, 32); }
    const int wid = threadIdx.x >> 5;
    if ((threadIdx.x & 31) == 0) { shm[wid] = s; shm[8 + wid] = q; }
    __syncthreads();
    if (threadIdx.x == 0) {
        float S = 0.f, Q = 0.f;
#pragma unroll
        for (int i = 0; i < 8; ++i) { S += shm[i]; Q += shm[8 + i]; }
        shm[16] = S; shm[17] = Q;
    }
    __syncthreads();
    s = shm[16]; q = shm[17];
}

// ---------------------------------------------------------------------------
// LayerNorm rows of width 512 (word-embedding stage)
// ---------------------------------------------------------------------------
__global__ __launch_bounds__(256)
void ln_rows_512(const float* __restrict__ X, const float* __restrict__ g,
                 const float* __restrict__ b, float* __restrict__ Y)
{
    __shared__ float shm[18];
    const int row = blockIdx.x;
    const int t = threadIdx.x;
    const float* x = X + (size_t)row * 512;
    const float v0 = x[t], v1 = x[t + 256];
    float s = v0 + v1, q = v0 * v0 + v1 * v1;
    blockReduce2(s, q, shm);
    const float mu  = s * (1.0f / 512.0f);
    const float inv = rsqrtf(q * (1.0f / 512.0f) - mu * mu + 1e-5f);
    Y[(size_t)row * 512 + t]       = (v0 - mu) * inv * g[t] + b[t];
    Y[(size_t)row * 512 + t + 256] = (v1 - mu) * inv * g[t + 256] + b[t + 256];
}

// ---------------------------------------------------------------------------
// Threefry-2x32 (JAX-compatible) for the Gumbel merge selection
// ---------------------------------------------------------------------------
__device__ __forceinline__ unsigned rotl32_(unsigned v, int r) { return (v << r) | (v >> (32 - r)); }

__device__ void threefry2x32_(unsigned k0, unsigned k1, unsigned x0, unsigned x1,
                              unsigned& o0, unsigned& o1)
{
    const unsigned ks2 = k0 ^ k1 ^ 0x1BD11BDAu;
    x0 += k0; x1 += k1;
    const int RA[4] = {13, 15, 26, 6};
    const int RB[4] = {17, 29, 16, 24};
#pragma unroll
    for (int i = 0; i < 4; ++i) { x0 += x1; x1 = rotl32_(x1, RA[i]); x1 ^= x0; }
    x0 += k1;  x1 += ks2 + 1u;
#pragma unroll
    for (int i = 0; i < 4; ++i) { x0 += x1; x1 = rotl32_(x1, RB[i]); x1 ^= x0; }
    x0 += ks2; x1 += k0 + 2u;
#pragma unroll
    for (int i = 0; i < 4; ++i) { x0 += x1; x1 = rotl32_(x1, RA[i]); x1 ^= x0; }
    x0 += k0;  x1 += k1 + 3u;
#pragma unroll
    for (int i = 0; i < 4; ++i) { x0 += x1; x1 = rotl32_(x1, RB[i]); x1 ^= x0; }
    x0 += k1;  x1 += ks2 + 4u;
#pragma unroll
    for (int i = 0; i < 4; ++i) { x0 += x1; x1 = rotl32_(x1, RA[i]); x1 ^= x0; }
    x0 += ks2; x1 += k0 + 5u;
    o0 = x0; o1 = x1;
}

__device__ float jax_uniform_(unsigned k0, unsigned k1, int e, int half, int ntot)
{
    const int j = (e < half) ? e : (e - half);
    const unsigned x0 = (unsigned)j;
    const unsigned x1 = ((half + j) < ntot) ? (unsigned)(half + j) : 0u;  // odd-count pad
    unsigned o0, o1;
    threefry2x32_(k0, k1, x0, x1, o0, o1);
    const unsigned bits = (e < half) ? o0 : o1;
    return __uint_as_float(0x3f800000u | (bits >> 9)) - 1.0f;
}

// one thread per batch row n: argmax over Sc of masked gumbel softmax
__global__ void argmax_sel_kernel(int* __restrict__ selIdx, const float* __restrict__ mask,
                                  int iter, int Sc)
{
    const int n = threadIdx.x;
    if (n >= 32) return;
    unsigned k0, k1;
    threefry2x32_(0u, 42u, 0u, (unsigned)iter, k0, k1);   // fold_in(key(42), iter)
    const int ntot = 32 * Sc;
    const int half = (ntot + 1) >> 1;
    float gmax = -1e30f;
    for (int s = 0; s < Sc; ++s) {
        const float u = jax_uniform_(k0, k1, n * Sc + s, half, ntot);
        const float gn = -logf(-logf(u + 1e-20f) + 1e-20f);
        gmax = fmaxf(gmax, gn);
    }
    float best = -1e30f; int bi = 0;
    for (int s = 0; s < Sc; ++s) {
        const float u = jax_uniform_(k0, k1, n * Sc + s, half, ntot);
        const float gn = -logf(-logf(u + 1e-20f) + 1e-20f);
        const float p = expf(gn - gmax) * mask[n * 48 + iter + 1 + s] + 1e-20f;
        if (p > best) { best = p; bi = s; }
    }
    selIdx[n] = bi;
}

// ---------------------------------------------------------------------------
// Fused gates + LayerNorm + tree-select + done-mask. One block per (s, n).
// ---------------------------------------------------------------------------
__global__ __launch_bounds__(256)
void combine_cell(const float* __restrict__ contents,   // M x 2048 (f1,f2,ig,parent)
                  const float* __restrict__ stateCur,   // (32*(Sc+1)) x 512
                  float* __restrict__ stateNext,        // (32*Sc) x 512
                  const int* __restrict__ selIdx,
                  const float* __restrict__ mask,       // 32 x 48
                  const float* __restrict__ g, const float* __restrict__ b,
                  int Sc, int iter, int lastFlag)
{
    __shared__ float shm[18];
    const int s = blockIdx.x, n = blockIdx.y;
    const int m = n * Sc + s;
    const float* c = contents + (size_t)m * 2048;
    const float* l = stateCur + (size_t)(n * (Sc + 1) + s) * 512;
    const float* r = l + 512;
    const int tid = threadIdx.x;

    float tval[2], lv[2], rv[2];
#pragma unroll
    for (int e = 0; e < 2; ++e) {
        const int d = tid + e * 256;
        lv[e] = l[d]; rv[e] = r[d];
        const float f1 = sigmoidf_(c[d]);
        const float f2 = sigmoidf_(c[512 + d]);
        const float ig = sigmoidf_(c[1024 + d]);
        tval[e] = f1 * lv[e] + f2 * rv[e] + ig * c[1536 + d];
    }
    float sum = tval[0] + tval[1];
    float sq  = tval[0] * tval[0] + tval[1] * tval[1];
    blockReduce2(sum, sq, shm);
    const float mu  = sum * (1.0f / 512.0f);
    const float inv = rsqrtf(sq * (1.0f / 512.0f) - mu * mu + 1e-5f);

    const int   k    = lastFlag ? 0 : selIdx[n];
    const float done = mask[n * 48 + iter + 1];
#pragma unroll
    for (int e = 0; e < 2; ++e) {
        const int d = tid + e * 256;
        const float ln = (tval[e] - mu) * inv * g[d] + b[d];
        float cand;
        if (lastFlag)      cand = ln;
        else if (s == k)   cand = ln;
        else if (s < k)    cand = lv[e];
        else               cand = rv[e];
        stateNext[(size_t)m * 512 + d] = done * cand + (1.0f - done) * lv[e];
    }
}

// ---------------------------------------------------------------------------
// Small utility kernels
// ---------------------------------------------------------------------------
__global__ void conv_transpose_bf16(const float* __restrict__ W, __bf16* __restrict__ WT,
                                    int K, int Nn)
{
    const int idx = blockIdx.x * 256 + threadIdx.x;
    if (idx >= K * Nn) return;
    const int n = idx / K, k = idx - n * K;
    WT[idx] = (__bf16)W[(size_t)k * Nn + n];     // WT[n][k] = W[k][n]
}

__global__ void cvt_f32_bf16(const float* __restrict__ X, __bf16* __restrict__ Y, int total)
{
    const int idx = blockIdx.x * 256 + threadIdx.x;
    if (idx < total) Y[idx] = (__bf16)X[idx];
}

// build x = concat(l, r) in bf16: row m=(n,s) -> [state[n][s], state[n][s+1]]
__global__ void pack_x_kernel(const float* __restrict__ state, __bf16* __restrict__ xbf,
                              int Sc, int total)
{
    const int idx = blockIdx.x * 256 + threadIdx.x;
    if (idx >= total) return;
    const int m = idx >> 10, cc = idx & 1023;
    const int n = m / Sc, s = m - n * Sc;
    const int srow = n * (Sc + 1) + s + (cc >> 9);
    xbf[idx] = (__bf16)state[(size_t)srow * 512 + (cc & 511)];
}

__global__ void copy_out_kernel(const float* __restrict__ state, float* __restrict__ out, int total)
{
    const int idx = blockIdx.x * 256 + threadIdx.x;
    if (idx < total) out[idx] = state[idx];   // final state: (32*1) x 512, rows == batch
}

// ---------------------------------------------------------------------------
// Host orchestration
// ---------------------------------------------------------------------------
extern "C" void kernel_launch(void* const* d_in, const int* in_sizes, int n_in,
                              void* d_out, int out_size, void* d_ws, size_t ws_size,
                              hipStream_t stream)
{
    (void)in_sizes; (void)n_in; (void)out_size; (void)ws_size;

    const float* input  = (const float*)d_in[0];   // 32x48x512
    const float* mask   = (const float*)d_in[1];   // 32x48
    const float* w_word = (const float*)d_in[2];   // 512x512
    const float* b_word = (const float*)d_in[3];
    const float* ln_g   = (const float*)d_in[4];
    const float* ln_b   = (const float*)d_in[5];
    const float* w1     = (const float*)d_in[6];   // 1024x2048
    const float* b1     = (const float*)d_in[7];
    const float* w2     = (const float*)d_in[8];   // 2048x2048
    const float* b2     = (const float*)d_in[9];
    const float* ln2_g  = (const float*)d_in[10];
    const float* ln2_b  = (const float*)d_in[11];
    float* out = (float*)d_out;

    char* ws = (char*)d_ws;
    auto alloc = [&](size_t bytes) { char* p = ws; ws += (bytes + 255) & ~(size_t)255; return p; };

    const int MMAX = 32 * 47;  // 1504
    float*  stateA   = (float*)alloc((size_t)32 * 48 * 512 * 4);
    float*  stateB   = (float*)alloc((size_t)32 * 48 * 512 * 4);
    float*  contents = (float*)alloc((size_t)MMAX * 2048 * 4);      // also word-GEMM y
    __bf16* xbf      = (__bf16*)alloc((size_t)MMAX * 1024 * 2);     // also input-bf16
    __bf16* interbf  = (__bf16*)alloc((size_t)MMAX * 2048 * 2);
    __bf16* wwT      = (__bf16*)alloc((size_t)512 * 512 * 2);
    __bf16* w1T      = (__bf16*)alloc((size_t)2048 * 1024 * 2);
    __bf16* w2T      = (__bf16*)alloc((size_t)2048 * 2048 * 2);
    int*    selIdx   = (int*)alloc(128);

    // One-time weight convert+transpose to bf16 (N-major for the B operand)
    conv_transpose_bf16<<<(512 * 512 + 255) / 256, 256, 0, stream>>>(w_word, wwT, 512, 512);
    conv_transpose_bf16<<<(1024 * 2048 + 255) / 256, 256, 0, stream>>>(w1, w1T, 1024, 2048);
    conv_transpose_bf16<<<(2048 * 2048 + 255) / 256, 256, 0, stream>>>(w2, w2T, 2048, 2048);

    // state0 = LN(input @ w_word + b_word)
    cvt_f32_bf16<<<(32 * 48 * 512 + 255) / 256, 256, 0, stream>>>(input, xbf, 32 * 48 * 512);
    {
        const int M0 = 32 * 48;  // 1536
        dim3 grid((M0 + BM - 1) / BM, 512 / BN);
        gemm_bf16_wmma<2><<<grid, 256, 0, stream>>>(xbf, 512, wwT, 512, b_word,
                                                    (void*)contents, 512, M0, 512);
        ln_rows_512<<<M0, 256, 0, stream>>>(contents, ln_g, ln_b, stateA);
    }

    float* cur = stateA;
    float* nxt = stateB;
    const int S = 48;
    for (int i = 0; i < S - 1; ++i) {
        const int curLen = S - i;
        const int Sc = curLen - 1;
        const int M = 32 * Sc;
        const int last = (i == S - 2) ? 1 : 0;

        pack_x_kernel<<<(M * 1024 + 255) / 256, 256, 0, stream>>>(cur, xbf, Sc, M * 1024);

        dim3 grid1((M + BM - 1) / BM, 2048 / BN);
        gemm_bf16_wmma<1><<<grid1, 256, 0, stream>>>(xbf, 1024, w1T, 1024, b1,
                                                     (void*)interbf, 2048, M, 1024);
        gemm_bf16_wmma<2><<<grid1, 256, 0, stream>>>(interbf, 2048, w2T, 2048, b2,
                                                     (void*)contents, 2048, M, 2048);

        if (!last)
            argmax_sel_kernel<<<1, 32, 0, stream>>>(selIdx, mask, i, Sc);

        dim3 gridc(Sc, 32);
        combine_cell<<<gridc, 256, 0, stream>>>(contents, cur, nxt, selIdx, mask,
                                                ln2_g, ln2_b, Sc, i, last);
        float* t = cur; cur = nxt; nxt = t;
    }

    copy_out_kernel<<<(32 * 512 + 255) / 256, 256, 0, stream>>>(cur, out, 32 * 512);
}
